// RegionAttn_29231547416952
// MI455X (gfx1250) — matrix operand order repaired and checked by reading, not compiled, not run on hardware
//
#include <hip/hip_runtime.h>

// ---------------------------------------------------------------------------
// CDNA5 (gfx1250) fused region-attention transformer block.
// bf16 WMMA (v_wmma_f32_16x16x32_bf16) for all GEMM-shaped work,
// async-to-LDS / TDM for tile movement.
// ---------------------------------------------------------------------------

typedef __attribute__((ext_vector_type(16))) __bf16 bf16x16;
typedef __attribute__((ext_vector_type(8)))  __bf16 bf16x8;
typedef __attribute__((ext_vector_type(8)))  float  f32x8;
typedef __attribute__((ext_vector_type(4)))  unsigned int u32x4;
typedef __attribute__((ext_vector_type(8)))  int    i32x8;
typedef __attribute__((ext_vector_type(4)))  int    i32x4;

#define AS1 __attribute__((address_space(1)))
#define AS3 __attribute__((address_space(3)))

#define DM   256      // d_model
#define DFF  1024     // d_ff
#define NC   19       // channels
#define NH   8        // heads
#define DK   32       // d_k = d_v
#define BSZ  4096     // batch
#define EPS  1e-5f

__constant__ int   c_rid[NC]      = {0,0,1,2,3,0,3,3,0,0,0,1,2,3,0,3,3,1,2};
__constant__ int   c_regch[4][7]  = {{0,1,5,8,9,10,14},{2,11,17,0,0,0,0},
                                     {3,12,18,0,0,0,0},{4,6,7,13,15,16,0}};
__constant__ int   c_regcnt[4]    = {7,3,3,6};
__constant__ float c_reginv[4]    = {1.f/7.f,1.f/3.f,1.f/3.f,1.f/6.f};

// ----- scalar bf16 helpers: native converts (v_cvt on gfx1250) -------------
__device__ __forceinline__ __bf16 f2bf(float f) { return (__bf16)f; }
__device__ __forceinline__ float  bf2f(__bf16 b) { return (float)b; }

// ----- async global -> LDS (ASYNCcnt path), fallback = reg copy ------------
__device__ __forceinline__ void async_b128(void* lds, const void* g) {
#if __has_builtin(__builtin_amdgcn_global_load_async_to_lds_b128)
    __builtin_amdgcn_global_load_async_to_lds_b128(
        (AS1 i32x4*)g, (AS3 i32x4*)lds, 0, 0);
#else
    *(u32x4*)lds = *(const u32x4*)g;
#endif
}
__device__ __forceinline__ void wait_async() {
#if __has_builtin(__builtin_amdgcn_s_wait_asynccnt)
    __builtin_amdgcn_s_wait_asynccnt(0);
#elif __has_builtin(__builtin_amdgcn_global_load_async_to_lds_b128)
    asm volatile("s_wait_asynccnt 0x0" ::: "memory");
#endif
}
__device__ __forceinline__ void wait_tensor() {
#if __has_builtin(__builtin_amdgcn_s_wait_tensorcnt)
    __builtin_amdgcn_s_wait_tensorcnt(0);
#else
    asm volatile("s_wait_tensorcnt 0x0" ::: "memory");
#endif
}

// ----- TDM: contiguous tile from global -> LDS (one wave issues) -----------
// 1-row 2D tile, data_size = 2 bytes.  Guarded: falls back to async copies.
__device__ __forceinline__ bool tdm_load_tile(void* lds, const void* g,
                                              unsigned elems2B) {
#if __has_builtin(__builtin_amdgcn_tensor_load_to_lds)
    unsigned lds_off = (unsigned)(unsigned long long)(AS3 char*)lds;
    unsigned long long ga = (unsigned long long)g;
    u32x4 g0;
    g0[0] = 1u;                                        // count=1, user mode
    g0[1] = lds_off;                                   // lds_addr
    g0[2] = (unsigned)(ga & 0xffffffffu);              // global_addr[31:0]
    g0[3] = (unsigned)((ga >> 32) & 0x01ffffffu)       // global_addr[56:32]
          | (2u << 30);                                // type = 2 (image)
    i32x8 g1 = {};
    g1[0] = (1 << 16);                                 // data_size = 2B
    g1[1] = (int)((elems2B & 0xffffu) << 16);          // tensor_dim0[15:0]
    g1[2] = (int)((elems2B >> 16) | (1u << 16));       // dim0 hi | tensor_dim1=1
    g1[3] = (int)(elems2B << 16);                      // tile_dim0
    g1[5] = (int)elems2B;                              // tensor_dim0_stride
    i32x4 gz = {};
#if __clang_major__ >= 23
    i32x8 gz8 = {};
    __builtin_amdgcn_tensor_load_to_lds(g0, g1, gz, gz, gz8, 0);
#else
    __builtin_amdgcn_tensor_load_to_lds(g0, g1, gz, gz, 0);
#endif
    return true;
#else
    (void)lds; (void)g; (void)elems2B;
    return false;
#endif
}

// ----- WMMA fragment movement ----------------------------------------------
__device__ __forceinline__ bf16x16 frag_a(const __bf16* base, int ld, int lane) {
    int row = lane & 15, hi = lane >> 4;
    const __bf16* p = base + row * ld + hi * 16;
    bf16x8 lo = *(const bf16x8*)(p);
    bf16x8 hh = *(const bf16x8*)(p + 8);
    bf16x16 f;
#pragma unroll
    for (int i = 0; i < 8; ++i) { f[i] = lo[i]; f[8 + i] = hh[i]; }
    return f;
}
__device__ __forceinline__ bf16x16 frag_b_nk(const __bf16* base, int ld, int lane) {
    int n = lane & 15, hi = lane >> 4;
    return *(const bf16x16*)(base + n * ld + hi * 16);
}
__device__ __forceinline__ bf16x16 frag_b_kn(const __bf16* base, int ld, int lane) {
    int n = lane & 15, hi = lane >> 4;
    bf16x16 f;
#pragma unroll
    for (int i = 0; i < 16; ++i) f[i] = base[(hi * 16 + i) * ld + n];
    return f;
}
__device__ __forceinline__ f32x8 wmma_bf16(bf16x16 a, bf16x16 b, f32x8 c) {
    return __builtin_amdgcn_wmma_f32_16x16x32_bf16(false, a, false, b,
                                                   (short)0, c, false, false);
}
__device__ __forceinline__ void acc_st_f32(float* base, int ld, int lane,
                                           f32x8 acc, float scale) {
    int n = lane & 15, hi = lane >> 4;
#pragma unroll
    for (int i = 0; i < 8; ++i) base[(hi * 8 + i) * ld + n] = acc[i] * scale;
}
__device__ __forceinline__ void acc_st_bf16(__bf16* base, int ld, int lane, f32x8 acc) {
    int n = lane & 15, hi = lane >> 4;
#pragma unroll
    for (int i = 0; i < 8; ++i) base[(hi * 8 + i) * ld + n] = f2bf(acc[i]);
}
__device__ __forceinline__ void acc_st_bf16_rl(__bf16* base, int ld, int lane,
                                               f32x8 acc, int mlimit) {
    int n = lane & 15, hi = lane >> 4;
#pragma unroll
    for (int i = 0; i < 8; ++i) {
        int m = hi * 8 + i;
        if (m < mlimit) base[m * ld + n] = f2bf(acc[i]);
    }
}
__device__ __forceinline__ void lds_fence() {
    asm volatile("s_wait_dscnt 0" ::: "memory");
}

// ---------------------------------------------------------------------------
// K0: f32 weight [K][N] -> bf16 transposed [N][K]
// ---------------------------------------------------------------------------
__global__ void k_convert_t(const float* __restrict__ W, __bf16* __restrict__ Wt,
                            int K, int N) {
    int total = K * N;
    for (int i = blockIdx.x * blockDim.x + threadIdx.x; i < total;
         i += gridDim.x * blockDim.x) {
        int k = i / N, n = i % N;
        Wt[n * K + k] = f2bf(W[i]);
    }
}

// ---------------------------------------------------------------------------
// K1: region pool + 4-token region self-attention + LN + scatter-add.
// ---------------------------------------------------------------------------
__global__ void __launch_bounds__(256)
k_region_attn(const float* __restrict__ x,
              const __bf16* __restrict__ wq, const __bf16* __restrict__ wk,
              const __bf16* __restrict__ wv, const __bf16* __restrict__ wo,
              __bf16* __restrict__ x1g) {
    extern __shared__ char smem[];
    __bf16* xs = (__bf16*)(smem);              // [8][19][256]  77824 B
    __bf16* fe = (__bf16*)(smem + 77824);      // [32][256]     16384 B
    __bf16* qs = (__bf16*)(smem + 94208);
    __bf16* ks = (__bf16*)(smem + 110592);
    __bf16* vs = (__bf16*)(smem + 126976);
    __bf16* cs = (__bf16*)(smem + 143360);
    float*  ot = (float* )(smem + 159744);     // [32][256] f32  32768 B

    const int tid = threadIdx.x, lane = tid & 31, wave = tid >> 5;
    const int b0 = blockIdx.x * 8;

    // ---- load x tile (f32 -> bf16, native cvt) ----
    const float4* xg4 = (const float4*)(x + (size_t)b0 * NC * DM);
    for (int i = tid; i < 8 * NC * DM / 4; i += 256) {
        float4 v = xg4[i];
        int e = i * 4;
        xs[e] = f2bf(v.x); xs[e+1] = f2bf(v.y); xs[e+2] = f2bf(v.z); xs[e+3] = f2bf(v.w);
    }
    __syncthreads();

    // ---- region mean pooling ----
    for (int e = tid; e < 32 * DM; e += 256) {
        int row = e >> 8, d = e & 255;
        int b = row >> 2, r = row & 3;
        float s = 0.f;
        for (int c = 0; c < c_regcnt[r]; ++c)
            s += bf2f(xs[(b * NC + c_regch[r][c]) * DM + d]);
        fe[row * DM + d] = f2bf(s * c_reginv[r]);
    }
    __syncthreads();

    // ---- Q/K/V projections: [32,256] x [256,256] ----
    for (int t = wave; t < 96; t += 8) {
        int proj = t / 32, mt = (t >> 4) & 1, nt = t & 15;
        const __bf16* W = proj == 0 ? wq : (proj == 1 ? wk : wv);
        __bf16* dst = proj == 0 ? qs : (proj == 1 ? ks : vs);
        f32x8 acc = {};
#pragma unroll
        for (int kt = 0; kt < 8; ++kt) {
            bf16x16 a = frag_a(fe + mt * 16 * DM + kt * 32, DM, lane);
            bf16x16 b = frag_b_nk(W + nt * 16 * DM + kt * 32, DM, lane);
            acc = wmma_bf16(a, b, acc);
        }
        acc_st_bf16(dst + mt * 16 * DM + nt * 16, DM, lane, acc);
    }
    __syncthreads();

    // ---- tiny 4x4 attention per (batch, head) ----
    if (tid < 64) {
        int b = tid >> 3, h = tid & 7, rb = b * 4, co = h * DK;
        float sc[4][4];
        for (int i = 0; i < 4; ++i)
            for (int j = 0; j < 4; ++j) {
                float s = 0.f;
                for (int k = 0; k < DK; ++k)
                    s += bf2f(qs[(rb+i)*DM + co + k]) * bf2f(ks[(rb+j)*DM + co + k]);
                sc[i][j] = s * 0.17677669529663689f;
            }
        for (int i = 0; i < 4; ++i) {
            float m = fmaxf(fmaxf(sc[i][0], sc[i][1]), fmaxf(sc[i][2], sc[i][3]));
            float s = 0.f;
            for (int j = 0; j < 4; ++j) { sc[i][j] = __expf(sc[i][j] - m); s += sc[i][j]; }
            float inv = 1.f / s;
            for (int v = 0; v < DK; ++v) {
                float a = 0.f;
                for (int j = 0; j < 4; ++j)
                    a += sc[i][j] * bf2f(vs[(rb+j)*DM + co + v]);
                cs[(rb+i)*DM + co + v] = f2bf(a * inv);
            }
        }
    }
    __syncthreads();

    // ---- output projection ----
    for (int t = wave; t < 32; t += 8) {
        int mt = t >> 4, nt = t & 15;
        f32x8 acc = {};
#pragma unroll
        for (int kt = 0; kt < 8; ++kt) {
            bf16x16 a = frag_a(cs + mt * 16 * DM + kt * 32, DM, lane);
            bf16x16 b = frag_b_nk(wo + nt * 16 * DM + kt * 32, DM, lane);
            acc = wmma_bf16(a, b, acc);
        }
        acc_st_f32(ot + mt * 16 * DM + nt * 16, DM, lane, acc, 1.f);
    }
    __syncthreads();

    // ---- LayerNorm(ctx@Wo + fe) -> refined (reuse qs) ----
    if (tid < 32) {
        int r = tid;
        float mu = 0.f;
        for (int d = 0; d < DM; ++d) mu += ot[r*DM+d] + bf2f(fe[r*DM+d]);
        mu *= (1.f / DM);
        float var = 0.f;
        for (int d = 0; d < DM; ++d) {
            float v = ot[r*DM+d] + bf2f(fe[r*DM+d]) - mu;
            var += v * v;
        }
        float inv = rsqrtf(var * (1.f / DM) + EPS);
        for (int d = 0; d < DM; ++d)
            qs[r*DM+d] = f2bf((ot[r*DM+d] + bf2f(fe[r*DM+d]) - mu) * inv);
    }
    __syncthreads();

    // ---- scatter-add -> x1 (streaming store) ----
    __bf16* outg = x1g + (size_t)b0 * NC * DM;
    for (int i = tid; i < 8 * NC * DM; i += 256) {
        int b = i / (NC * DM), rem = i % (NC * DM);
        int c = rem >> 8, d = rem & 255;
        float v = bf2f(xs[i]) + bf2f(qs[((b << 2) + c_rid[c]) * DM + d]);
        __builtin_nontemporal_store(f2bf(v), outg + i);
    }
}

// ---------------------------------------------------------------------------
// K2: full 19-token MHA + LN.  4 batch elements / block (80 padded rows).
// ---------------------------------------------------------------------------
__global__ void __launch_bounds__(256)
k_lsa(const __bf16* __restrict__ x1g,
      const __bf16* __restrict__ wq, const __bf16* __restrict__ wk,
      const __bf16* __restrict__ wv, const __bf16* __restrict__ wo,
      __bf16* __restrict__ fusedg) {
    extern __shared__ char smem[];
    __bf16* a  = (__bf16*)(smem);              // [80][256]  40960 B
    __bf16* qs = (__bf16*)(smem + 40960);
    __bf16* ks = (__bf16*)(smem + 81920);
    __bf16* vs = (__bf16*)(smem + 122880);
    __bf16* cs = (__bf16*)(smem + 163840);
    float*  sw = (float* )(smem + 204800);     // [8 waves][32][32] f32
    __bf16* aw = (__bf16*)(smem + 237568);     // [8 waves][32][32] bf16
    float*  ot = (float*)qs;                   // reuse qs+ks after attention

    const int tid = threadIdx.x, lane = tid & 31, wave = tid >> 5;
    const int b0 = blockIdx.x * 4;
    const size_t gbase = (size_t)b0 * NC * DM;

    // ---- async-load 76 rows into LDS, zero-pad to 80 ----
    {
        const __bf16* src = x1g + gbase;
        for (int i = tid; i < 76 * DM / 8; i += 256)
            async_b128(a + i * 8, src + i * 8);
        for (int i = tid; i < 4 * DM; i += 256) a[76 * DM + i] = f2bf(0.f);
        wait_async();
    }
    __syncthreads();

    // ---- Q/K/V projections: [80,256] x [256,256] ----
    for (int t = wave; t < 240; t += 8) {
        int proj = t / 80, r = t % 80, mt = r / 16, nt = r % 16;
        const __bf16* W = proj == 0 ? wq : (proj == 1 ? wk : wv);
        __bf16* dst = proj == 0 ? qs : (proj == 1 ? ks : vs);
        f32x8 acc = {};
#pragma unroll
        for (int kt = 0; kt < 8; ++kt) {
            bf16x16 af = frag_a(a + mt * 16 * DM + kt * 32, DM, lane);
            bf16x16 bf = frag_b_nk(W + nt * 16 * DM + kt * 32, DM, lane);
            acc = wmma_bf16(af, bf, acc);
        }
        acc_st_bf16(dst + mt * 16 * DM + nt * 16, DM, lane, acc);
    }
    __syncthreads();

    // ---- attention: 32 (batch, head) pairs, 4 per wave ----
    float*  swv = sw + wave * 32 * 32;
    __bf16* awv = aw + wave * 32 * 32;
    for (int i = 0; i < 4; ++i) {
        int p = wave * 4 + i, b = p >> 3, h = p & 7;
        int qrow = b * NC, co = h * DK;
#pragma unroll
        for (int mt = 0; mt < 2; ++mt)
#pragma unroll
            for (int nt = 0; nt < 2; ++nt) {
                f32x8 acc = {};
                bf16x16 af = frag_a(qs + (qrow + mt * 16) * DM + co, DM, lane);
                bf16x16 bf = frag_b_nk(ks + (qrow + nt * 16) * DM + co, DM, lane);
                acc = wmma_bf16(af, bf, acc);
                int n = lane & 15, hi = lane >> 4;
#pragma unroll
                for (int e = 0; e < 8; ++e)
                    swv[(mt*16 + hi*8 + e) * 32 + nt*16 + n] =
                        acc[e] * 0.17677669529663689f;
            }
        lds_fence();
        {
            int r = lane;
            float m = -1e30f;
            for (int j = 0; j < NC; ++j) m = fmaxf(m, swv[r*32+j]);
            float s = 0.f;
            for (int j = 0; j < NC; ++j) { float e = __expf(swv[r*32+j] - m); swv[r*32+j] = e; s += e; }
            float inv = 1.f / s;
            for (int j = 0; j < 32; ++j)
                awv[r*32+j] = f2bf(j < NC ? swv[r*32+j] * inv : 0.f);
        }
        lds_fence();
#pragma unroll
        for (int mt = 0; mt < 2; ++mt)
#pragma unroll
            for (int nt = 0; nt < 2; ++nt) {
                f32x8 acc = {};
                bf16x16 af = frag_a(awv + mt * 16 * 32, 32, lane);
                bf16x16 bf = frag_b_kn(vs + qrow * DM + co + nt * 16, DM, lane);
                acc = wmma_bf16(af, bf, acc);
                acc_st_bf16_rl(cs + (qrow + mt * 16) * DM + co + nt * 16, DM, lane,
                               acc, NC - mt * 16);
            }
        lds_fence();
    }
    __syncthreads();

    // ---- output projection ----
    for (int t = wave; t < 80; t += 8) {
        int mt = t / 16, nt = t % 16;
        f32x8 acc = {};
#pragma unroll
        for (int kt = 0; kt < 8; ++kt) {
            bf16x16 af = frag_a(cs + mt * 16 * DM + kt * 32, DM, lane);
            bf16x16 bf = frag_b_nk(wo + nt * 16 * DM + kt * 32, DM, lane);
            acc = wmma_bf16(af, bf, acc);
        }
        acc_st_f32(ot + mt * 16 * DM + nt * 16, DM, lane, acc, 1.f);
    }
    __syncthreads();

    // ---- LayerNorm(ctx@Wo + x1) -> fused ----
    if (tid < 76) {
        int r = tid;
        float mu = 0.f;
        for (int d = 0; d < DM; ++d) mu += ot[r*DM+d] + bf2f(a[r*DM+d]);
        mu *= (1.f / DM);
        float var = 0.f;
        for (int d = 0; d < DM; ++d) {
            float v = ot[r*DM+d] + bf2f(a[r*DM+d]) - mu;
            var += v * v;
        }
        float inv = rsqrtf(var * (1.f / DM) + EPS);
        __bf16* dst = fusedg + gbase + (size_t)r * DM;
        for (int d = 0; d < DM; ++d)
            __builtin_nontemporal_store(
                f2bf((ot[r*DM+d] + bf2f(a[r*DM+d]) - mu) * inv), dst + d);
    }
}

// ---------------------------------------------------------------------------
// K3: FFN (relu(fused@W1)@W2) + residual + final LN.  64 rows / block.
// ---------------------------------------------------------------------------
__global__ void __launch_bounds__(256)
k_ffn(const __bf16* __restrict__ fusedg,
      const __bf16* __restrict__ w1t,   // [1024][256] bf16
      const __bf16* __restrict__ w2t,   // [256][1024] bf16
      float* __restrict__ outg) {
    extern __shared__ char smem[];
    __bf16* fs = (__bf16*)(smem);            // [64][256]   32768 B
    __bf16* hs = (__bf16*)(smem + 32768);    // [64][1024] 131072 B
    float*  ot = (float* )(smem + 163840);   // [64][256]   65536 B

    const int tid = threadIdx.x, lane = tid & 31, wave = tid >> 5;
    const size_t r0 = (size_t)blockIdx.x * 64;

    // ---- fused tile: TDM descriptor load (wave 0), fallback = async ----
    {
        bool tdm = false;
        if (wave == 0) {
            tdm = tdm_load_tile(fs, fusedg + r0 * DM, 64 * DM);
            if (tdm) wait_tensor();
        }
#if !__has_builtin(__builtin_amdgcn_tensor_load_to_lds)
        const __bf16* src = fusedg + r0 * DM;
        for (int i = tid; i < 64 * DM / 8; i += 256)
            async_b128(fs + i * 8, src + i * 8);
        wait_async();
#endif
        (void)tdm;
    }
    __syncthreads();

    // ---- h = relu(fused @ W1): [64,256] x [256,1024] ----
    for (int t = wave; t < 256; t += 8) {
        int mt = t >> 6, nt = t & 63;
        const __bf16* bp = w1t + nt * 16 * DM;
        __builtin_prefetch(bp + 16 * DM, 0, 0);
        f32x8 acc = {};
#pragma unroll
        for (int kt = 0; kt < 8; ++kt) {
            bf16x16 af = frag_a(fs + mt * 16 * DM + kt * 32, DM, lane);
            bf16x16 bf = frag_b_nk(bp + kt * 32, DM, lane);
            acc = wmma_bf16(af, bf, acc);
        }
#pragma unroll
        for (int e = 0; e < 8; ++e) acc[e] = fmaxf(acc[e], 0.f);
        acc_st_bf16(hs + mt * 16 * DFF + nt * 16, DFF, lane, acc);
    }
    __syncthreads();

    // ---- o = h @ W2: [64,1024] x [1024,256] ----
    for (int t = wave; t < 64; t += 8) {
        int mt = t >> 4, nt = t & 15;
        f32x8 acc = {};
#pragma unroll 8
        for (int kt = 0; kt < 32; ++kt) {
            bf16x16 af = frag_a(hs + mt * 16 * DFF + kt * 32, DFF, lane);
            bf16x16 bf = frag_b_nk(w2t + nt * 16 * DFF + kt * 32, DFF, lane);
            acc = wmma_bf16(af, bf, acc);
        }
        acc_st_f32(ot + mt * 16 * DM + nt * 16, DM, lane, acc, 1.f);
    }
    __syncthreads();

    // ---- LayerNorm(ffn + fused) -> f32 output (streaming) ----
    if (tid < 64) {
        int r = tid;
        float mu = 0.f;
        for (int d = 0; d < DM; ++d) mu += ot[r*DM+d] + bf2f(fs[r*DM+d]);
        mu *= (1.f / DM);
        float var = 0.f;
        for (int d = 0; d < DM; ++d) {
            float v = ot[r*DM+d] + bf2f(fs[r*DM+d]) - mu;
            var += v * v;
        }
        float inv = rsqrtf(var * (1.f / DM) + EPS);
        float* dst = outg + (r0 + r) * DM;
        for (int d = 0; d < DM; ++d)
            __builtin_nontemporal_store(
                (ot[r*DM+d] + bf2f(fs[r*DM+d]) - mu) * inv, dst + d);
    }
}

// ---------------------------------------------------------------------------
extern "C" void kernel_launch(void* const* d_in, const int* in_sizes, int n_in,
                              void* d_out, int out_size, void* d_ws, size_t ws_size,
                              hipStream_t stream) {
    (void)in_sizes; (void)n_in; (void)out_size; (void)ws_size;
    const float* x      = (const float*)d_in[0];
    const float* rsa_w[4] = {(const float*)d_in[1], (const float*)d_in[2],
                             (const float*)d_in[3], (const float*)d_in[4]};
    const float* lsa_w[4] = {(const float*)d_in[5], (const float*)d_in[6],
                             (const float*)d_in[7], (const float*)d_in[8]};
    const float* ffn_w1 = (const float*)d_in[9];
    const float* ffn_w2 = (const float*)d_in[10];
    float* out = (float*)d_out;

    char* ws = (char*)d_ws;
    __bf16* wt[8];
    for (int i = 0; i < 8; ++i) wt[i] = (__bf16*)(ws + (size_t)i * 131072);
    __bf16* w1t   = (__bf16*)(ws + 1048576);          // [1024][256]
    __bf16* w2t   = (__bf16*)(ws + 1572864);          // [256][1024]
    __bf16* x1    = (__bf16*)(ws + 2097152);          // [B][19][256]
    __bf16* fused = (__bf16*)(ws + 41943040);         // [B][19][256]

    for (int i = 0; i < 4; ++i)
        k_convert_t<<<64, 256, 0, stream>>>(rsa_w[i], wt[i], DM, DM);
    for (int i = 0; i < 4; ++i)
        k_convert_t<<<64, 256, 0, stream>>>(lsa_w[i], wt[4 + i], DM, DM);
    k_convert_t<<<256, 256, 0, stream>>>(ffn_w1, w1t, DM, DFF);
    k_convert_t<<<256, 256, 0, stream>>>(ffn_w2, w2t, DFF, DM);

    k_region_attn<<<BSZ / 8, 256, 192512, stream>>>(
        x, wt[0], wt[1], wt[2], wt[3], x1);
    k_lsa<<<BSZ / 4, 256, 253952, stream>>>(
        x1, wt[4], wt[5], wt[6], wt[7], fused);
    k_ffn<<<BSZ * NC / 64, 256, 229376, stream>>>(fused, w1t, w2t, out);
}
